// Qwen3Attention_7653631721674
// MI455X (gfx1250) — matrix operand order repaired and checked by reading, not compiled, not run on hardware
//
#include <hip/hip_runtime.h>

typedef unsigned short u16;
typedef __attribute__((ext_vector_type(16))) __bf16 v16bf;
typedef __attribute__((ext_vector_type(8)))  float  v8f;

#define S_LEN 2048
#define HD    128
#define NH    16
#define NKV   8
#define DMODEL 1024
#define BSZ   2
#define MROWS (BSZ * S_LEN)          // 4096

struct U128x2 { uint4 a, b; };

static __device__ __forceinline__ v16bf make_frag(uint4 a, uint4 b) {
    U128x2 t{a, b};
    return __builtin_bit_cast(v16bf, t);
}

// float -> bf16 (round to nearest even)
static __device__ __forceinline__ u16 f2bf(float f) {
    unsigned u = __builtin_bit_cast(unsigned, f);
    u += 0x7FFFu + ((u >> 16) & 1u);
    return (u16)(u >> 16);
}

// ---------------------------------------------------------------------------
// elementwise f32 -> bf16 conversion
// ---------------------------------------------------------------------------
__global__ __launch_bounds__(256) void cvt_bf16(const float* __restrict__ x,
                                                u16* __restrict__ y, long long n) {
    long long i = (long long)blockIdx.x * 256 + threadIdx.x;
    if (i < n) y[i] = f2bf(x[i]);
}

// ---------------------------------------------------------------------------
// C[M,N] (f32) = A[M,K](bf16) @ B[N,K](bf16)^T      tiles: block 32x64, wave 16x16
// ---------------------------------------------------------------------------
__global__ __launch_bounds__(256) void gemm_bf16(const u16* __restrict__ A,
                                                 const u16* __restrict__ B,
                                                 float* __restrict__ C,
                                                 int M, int N, int K) {
    const int lane = threadIdx.x & 31, wv = threadIdx.x >> 5;
    const int hi = lane >> 4, l15 = lane & 15;
    const int tm = blockIdx.y * 32 + (wv >> 2) * 16;
    const int tn = blockIdx.x * 64 + (wv & 3) * 16;

    const u16* pa = A + (size_t)(tm + l15) * K + hi * 8;    // A: lane = row M
    const u16* pb = B + (size_t)(tn + l15) * K + hi * 8;    // B: lane = col N
    v8f acc = {0.f, 0.f, 0.f, 0.f, 0.f, 0.f, 0.f, 0.f};

    for (int k0 = 0; k0 < K; k0 += 32) {
        uint4 a0 = *(const uint4*)(pa);
        uint4 a1 = *(const uint4*)(pa + 16);
        uint4 b0 = *(const uint4*)(pb);
        uint4 b1 = *(const uint4*)(pb + 16);
        __builtin_prefetch(pa + 128, 0, 1);
        __builtin_prefetch(pb + 128, 0, 1);
        acc = __builtin_amdgcn_wmma_f32_16x16x32_bf16(
            false, make_frag(a0, a1), false, make_frag(b0, b1),
            (short)0, acc, false, false);
        pa += 32; pb += 32;
    }
    float* pc = C + (size_t)(tm + hi * 8) * N + tn + l15;   // C: VGPR j -> row j / j+8
#pragma unroll
    for (int j = 0; j < 8; ++j) pc[(size_t)j * N] = acc[j];
}

// ---------------------------------------------------------------------------
// RMSNorm + RoPE for q and k rows; writes head-major bf16 [b,h,s,hd]
// one wave per (row, head); lane holds d, d+32, d+64, d+96 (rotate pairs in-lane)
// ---------------------------------------------------------------------------
static __device__ __forceinline__ void normrope_row(const float* __restrict__ src,
                                                    const float* __restrict__ nw,
                                                    const float* __restrict__ cp,
                                                    const float* __restrict__ sp,
                                                    u16* __restrict__ dst, int lane) {
    const int d0 = lane, d1 = lane + 32, d2 = lane + 64, d3 = lane + 96;
    float x0 = src[d0], x1 = src[d1], x2 = src[d2], x3 = src[d3];
    float ss = x0 * x0 + x1 * x1 + x2 * x2 + x3 * x3;
    for (int m = 16; m; m >>= 1) ss += __shfl_xor(ss, m, 32);
    const float r = rsqrtf(ss * (1.0f / 128.0f) + 1e-6f);
    x0 *= r * nw[d0]; x1 *= r * nw[d1]; x2 *= r * nw[d2]; x3 *= r * nw[d3];
    float y0 = x0 * cp[d0] - x2 * sp[d0];
    float y1 = x1 * cp[d1] - x3 * sp[d1];
    float y2 = x2 * cp[d2] + x0 * sp[d2];
    float y3 = x3 * cp[d3] + x1 * sp[d3];
    dst[d0] = f2bf(y0); dst[d1] = f2bf(y1); dst[d2] = f2bf(y2); dst[d3] = f2bf(y3);
}

__global__ __launch_bounds__(256) void qk_norm_rope(const float* __restrict__ qraw,
                                                    const float* __restrict__ kraw,
                                                    const float* __restrict__ cosp,
                                                    const float* __restrict__ sinp,
                                                    const float* __restrict__ qw,
                                                    const float* __restrict__ kw,
                                                    u16* __restrict__ Qo,
                                                    u16* __restrict__ Ko) {
    const int lane = threadIdx.x & 31, wv = threadIdx.x >> 5;
    const long long g = (long long)blockIdx.x * 8 + wv;
    const long long NQ = (long long)MROWS * NH;
    const long long NK = (long long)MROWS * NKV;
    if (g < NQ) {
        int row = (int)(g >> 4), h = (int)(g & 15);
        int b = row >> 11, s = row & (S_LEN - 1);
        normrope_row(qraw + (size_t)row * (NH * HD) + h * HD, qw,
                     cosp + (size_t)row * HD, sinp + (size_t)row * HD,
                     Qo + ((size_t)(b * NH + h) * S_LEN + s) * HD, lane);
    } else if (g < NQ + NK) {
        long long gk = g - NQ;
        int row = (int)(gk >> 3), h = (int)(gk & 7);
        int b = row >> 11, s = row & (S_LEN - 1);
        normrope_row(kraw + (size_t)row * (NKV * HD) + h * HD, kw,
                     cosp + (size_t)row * HD, sinp + (size_t)row * HD,
                     Ko + ((size_t)(b * NKV + h) * S_LEN + s) * HD, lane);
    }
}

// ---------------------------------------------------------------------------
// V [b,s,kv,d] f32  ->  Vt [b,kv,d,s] bf16  (so P@V B-fragments load contiguously)
// ---------------------------------------------------------------------------
__global__ __launch_bounds__(256) void v_transpose(const float* __restrict__ vraw,
                                                   u16* __restrict__ Vt) {
    const int bkd = blockIdx.x;                 // (b*8+kv)*128 + d
    const int s = blockIdx.y * 256 + threadIdx.x;
    const int d = bkd & 127, bk = bkd >> 7;
    const int b = bk >> 3, kv = bk & 7;
    float v = vraw[((size_t)(b * S_LEN + s)) * (NKV * HD) + kv * HD + d];
    Vt[(size_t)bkd * S_LEN + s] = f2bf(v);
}

// ---------------------------------------------------------------------------
// Attention: block = 8 waves = 128 query rows of one (b,h); wave = 16 rows.
// K/V staged cooperatively into LDS with async loads (double buffered,
// 32-key granules, ASYNCcnt-tracked, overlapped with WMMA compute).
// Unnormalized exp-scores go straight to AW (safe: RMS-normed q,k bound
// |score| <= sqrt(128), so no max-subtraction needed); O normalized by
// row-sum in-kernel, row-sums exported for the AW normalize pass.
// ---------------------------------------------------------------------------
__global__ __launch_bounds__(256) void attn_fwd(const u16* __restrict__ Q,
                                                const u16* __restrict__ Kc,
                                                const u16* __restrict__ Vt,
                                                float* __restrict__ AW,
                                                u16* __restrict__ O,
                                                float* __restrict__ RS) {
    __shared__ __align__(16) u16 Ks[2][32][136];   // 32 keys x 128 hd (+8 pad)
    __shared__ __align__(16) u16 Vs[2][128][40];   // 128 d  x 32 keys (+8 pad)
    __shared__ __align__(16) u16 plds[8][16][40];  // per-wave P 16x32 (+8 pad)

    const int tid = threadIdx.x;
    const int lane = tid & 31, wv = tid >> 5;
    const int hi = lane >> 4, l15 = lane & 15;
    const int bh = blockIdx.y;                 // b*16 + h
    const int b = bh >> 4, h = bh & 15;
    const int kvh = h >> 1;
    const int q0 = blockIdx.x * 128 + wv * 16;
    const int gd = (q0 + 15) >> 5;             // wave's last 32-key granule
    const int gmaxb = blockIdx.x * 4 + 3;      // block-uniform loop bound
    const float scale = 0.08838834764831845f;  // 128^-0.5
    const v8f vzero = {0.f, 0.f, 0.f, 0.f, 0.f, 0.f, 0.f, 0.f};

    // Q fragments for this wave's 16 rows (4 chunks of K=32 over hd=128)
    const u16* Qrow = Q + ((size_t)bh * S_LEN + q0 + l15) * HD;
    v16bf qf[4];
#pragma unroll
    for (int c = 0; c < 4; ++c) {
        const u16* p = Qrow + c * 32 + hi * 8;
        qf[c] = make_frag(*(const uint4*)p, *(const uint4*)(p + 16));
    }
    const u16* KbG = Kc + (size_t)(b * NKV + kvh) * S_LEN * HD;
    const u16* VbG = Vt + (size_t)(b * NKV + kvh) * HD * S_LEN;

    v8f oacc[8];
#pragma unroll
    for (int dt = 0; dt < 8; ++dt) oacc[dt] = vzero;
    float rsj[8];
#pragma unroll
    for (int j = 0; j < 8; ++j) rsj[j] = 0.f;

    // cooperative async staging of one 32-key K/V granule (4 async ops/wave)
    auto stage = [&](int g, int buf) {
        // K granule: 8 KB = 512 x 16B chunks; chunk c: key=c>>3, off=(c&7)*8
#pragma unroll
        for (int it = 0; it < 2; ++it) {
            int c = tid + it * 256;
            int key = c >> 3, off = (c & 7) * 8;
            const u16* gp = KbG + (size_t)(g * 32 + key) * HD + off;
            unsigned la = (unsigned)(uintptr_t)&Ks[buf][key][off];
            asm volatile("global_load_async_to_lds_b128 %0, %1, off"
                         :: "v"(la), "v"(gp) : "memory");
        }
        // V granule: 8 KB = 512 x 16B chunks; chunk c: d=c>>2, off=(c&3)*8
#pragma unroll
        for (int it = 0; it < 2; ++it) {
            int c = tid + it * 256;
            int d = c >> 2, off = (c & 3) * 8;
            const u16* gp = VbG + (size_t)d * S_LEN + g * 32 + off;
            unsigned la = (unsigned)(uintptr_t)&Vs[buf][d][off];
            asm volatile("global_load_async_to_lds_b128 %0, %1, off"
                         :: "v"(la), "v"(gp) : "memory");
        }
    };

    stage(0, 0);
    for (int g = 0; g <= gmaxb; ++g) {
        const int buf = g & 1;
        if (g < gmaxb) {
            stage(g + 1, buf ^ 1);                       // prefetch next granule
            asm volatile("s_wait_asynccnt 0x4" ::: "memory");  // wait older 4
        } else {
            asm volatile("s_wait_asynccnt 0x0" ::: "memory");
        }
        __syncthreads();                                 // granule g visible to all

        if (g <= gd) {
            // two 16-key score subtiles:  S = Q(16xhd) . K^T(hd x 16)
#pragma unroll
            for (int t16 = 0; t16 < 2; ++t16) {
                v8f acc = vzero;
                const u16* kr = &Ks[buf][t16 * 16 + l15][0];
#pragma unroll
                for (int c = 0; c < 4; ++c) {
                    const u16* p = kr + c * 32 + hi * 8;
                    acc = __builtin_amdgcn_wmma_f32_16x16x32_bf16(
                        false, qf[c], false,
                        make_frag(*(const uint4*)p, *(const uint4*)(p + 16)),
                        (short)0, acc, false, false);
                }
                const int kcol = g * 32 + t16 * 16 + l15;
#pragma unroll
                for (int j = 0; j < 8; ++j) {
                    const int qrow = q0 + hi * 8 + j;
                    float p = (kcol <= qrow) ? __expf(acc[j] * scale) : 0.f;
                    AW[((size_t)bh * S_LEN + qrow) * S_LEN + kcol] = p;  // unnormalized
                    float t = p;                                          // row-sum
                    t += __shfl_xor(t, 1, 32);
                    t += __shfl_xor(t, 2, 32);
                    t += __shfl_xor(t, 4, 32);
                    t += __shfl_xor(t, 8, 32);
                    rsj[j] += t;
                    plds[wv][hi * 8 + j][t16 * 16 + l15] = f2bf(p);       // C->A layout
                }
            }
            // P@V over all 32 keys of the granule (K=32 A-fragment from LDS)
            uint4 u0 = *(const uint4*)&plds[wv][l15][hi * 8];
            uint4 u1 = *(const uint4*)&plds[wv][l15][hi * 8 + 16];
            v16bf pf = make_frag(u0, u1);
#pragma unroll
            for (int dt = 0; dt < 8; ++dt) {
                const u16* pv = &Vs[buf][dt * 16 + l15][hi * 8];
                oacc[dt] = __builtin_amdgcn_wmma_f32_16x16x32_bf16(
                    false, pf, false,
                    make_frag(*(const uint4*)pv, *(const uint4*)(pv + 16)),
                    (short)0, oacc[dt], false, false);
            }
        }
        __syncthreads();   // all waves done with buf before it is re-staged
    }

    // zero-fill the fully-masked (future) key region of AW
    {
        const float4 z4 = {0.f, 0.f, 0.f, 0.f};
        const int cstart = (gd + 1) * 32;
        for (int m = 0; m < 16; ++m) {
            float* rp = AW + ((size_t)bh * S_LEN + q0 + m) * S_LEN;
            for (int c = cstart + lane * 4; c < S_LEN; c += 128)
                *(float4*)(rp + c) = z4;
        }
    }

    // normalize O by row-sum, export row-sums, store O as [b,s, h*hd] bf16
#pragma unroll
    for (int j = 0; j < 8; ++j) {
        const int qrow = q0 + hi * 8 + j;
        const float inv = 1.0f / rsj[j];
        if (l15 == 0) RS[(size_t)bh * S_LEN + qrow] = rsj[j];
        u16* orow = O + ((size_t)(b * S_LEN + qrow)) * (NH * HD) + h * HD;
#pragma unroll
        for (int dt = 0; dt < 8; ++dt)
            orow[dt * 16 + l15] = f2bf(oacc[dt][j] * inv);
    }
}

// ---------------------------------------------------------------------------
// attn_weights /= rowsum  (float4 per thread; row uniform within a float4)
// ---------------------------------------------------------------------------
__global__ __launch_bounds__(256) void norm_attn(float* __restrict__ aw,
                                                 const float* __restrict__ rs,
                                                 long long n4) {
    long long i = (long long)blockIdx.x * 256 + threadIdx.x;
    if (i >= n4) return;
    float4 v = ((float4*)aw)[i];
    const float inv = 1.0f / rs[(i << 2) >> 11];   // flat/2048 = row index
    v.x *= inv; v.y *= inv; v.z *= inv; v.w *= inv;
    ((float4*)aw)[i] = v;
}

// ---------------------------------------------------------------------------
extern "C" void kernel_launch(void* const* d_in, const int* in_sizes, int n_in,
                              void* d_out, int out_size, void* d_ws, size_t ws_size,
                              hipStream_t stream) {
    const float* hs   = (const float*)d_in[0];
    const float* cosp = (const float*)d_in[1];
    const float* sinp = (const float*)d_in[2];
    // d_in[3]: attention_mask (causal, hardcoded)
    const float* Wq = (const float*)d_in[4];
    const float* Wk = (const float*)d_in[5];
    const float* Wv = (const float*)d_in[6];
    const float* Wo = (const float*)d_in[7];
    const float* qnw = (const float*)d_in[8];
    const float* knw = (const float*)d_in[9];

    char* w = (char*)d_ws;
    size_t off = 0;
    auto alloc = [&](size_t bytes) {
        void* p = w + off;
        off += (bytes + 255) & ~(size_t)255;
        return p;
    };
    u16* Xb   = (u16*)alloc((size_t)MROWS * DMODEL * 2);          // 8 MB
    u16* Wqb  = (u16*)alloc((size_t)NH * HD * DMODEL * 2);        // 4 MB
    u16* Wkb  = (u16*)alloc((size_t)NKV * HD * DMODEL * 2);       // 2 MB
    u16* Wvb  = (u16*)alloc((size_t)NKV * HD * DMODEL * 2);       // 2 MB
    u16* Wob  = (u16*)alloc((size_t)DMODEL * NH * HD * 2);        // 4 MB
    float* qraw = (float*)alloc((size_t)MROWS * NH * HD * 4);     // 32 MB
    float* kraw = (float*)alloc((size_t)MROWS * NKV * HD * 4);    // 16 MB
    float* vraw = (float*)alloc((size_t)MROWS * NKV * HD * 4);    // 16 MB
    u16* Qb   = (u16*)alloc((size_t)BSZ * NH * S_LEN * HD * 2);   // 16 MB
    u16* Kb   = (u16*)alloc((size_t)BSZ * NKV * S_LEN * HD * 2);  // 8 MB
    u16* Vtb  = (u16*)alloc((size_t)BSZ * NKV * HD * S_LEN * 2);  // 8 MB
    u16* Ob   = (u16*)alloc((size_t)MROWS * NH * HD * 2);         // 16 MB
    float* rsum = (float*)alloc((size_t)BSZ * NH * S_LEN * 4);    // 256 KB

    float* attn_out = (float*)d_out;                        // [B,S,1024]
    float* aw = (float*)d_out + (size_t)MROWS * DMODEL;     // [B,H,S,S]

    // 1) bf16 conversions
    cvt_bf16<<<(MROWS * DMODEL + 255) / 256, 256, 0, stream>>>(hs, Xb, (long long)MROWS * DMODEL);
    cvt_bf16<<<(NH * HD * DMODEL + 255) / 256, 256, 0, stream>>>(Wq, Wqb, (long long)NH * HD * DMODEL);
    cvt_bf16<<<(NKV * HD * DMODEL + 255) / 256, 256, 0, stream>>>(Wk, Wkb, (long long)NKV * HD * DMODEL);
    cvt_bf16<<<(NKV * HD * DMODEL + 255) / 256, 256, 0, stream>>>(Wv, Wvb, (long long)NKV * HD * DMODEL);
    cvt_bf16<<<(DMODEL * NH * HD + 255) / 256, 256, 0, stream>>>(Wo, Wob, (long long)DMODEL * NH * HD);

    // 2) QKV projections (WMMA)
    gemm_bf16<<<dim3((NH * HD) / 64, MROWS / 32), 256, 0, stream>>>(Xb, Wqb, qraw, MROWS, NH * HD, DMODEL);
    gemm_bf16<<<dim3((NKV * HD) / 64, MROWS / 32), 256, 0, stream>>>(Xb, Wkb, kraw, MROWS, NKV * HD, DMODEL);
    gemm_bf16<<<dim3((NKV * HD) / 64, MROWS / 32), 256, 0, stream>>>(Xb, Wvb, vraw, MROWS, NKV * HD, DMODEL);

    // 3) RMSNorm + RoPE (q,k) and V transpose
    qk_norm_rope<<<(MROWS * (NH + NKV)) / 8, 256, 0, stream>>>(qraw, kraw, cosp, sinp, qnw, knw, Qb, Kb);
    v_transpose<<<dim3(BSZ * NKV * HD, S_LEN / 256), 256, 0, stream>>>(vraw, Vtb);

    // 4) attention (WMMA + async LDS staging)
    attn_fwd<<<dim3(S_LEN / 128, BSZ * NH), 256, 0, stream>>>(Qb, Kb, Vtb, aw, Ob, rsum);

    // 5) normalize attn_weights by row-sums
    long long n4 = (long long)BSZ * NH * S_LEN * S_LEN / 4;
    norm_attn<<<(unsigned)(n4 / 256), 256, 0, stream>>>(aw, rsum, n4);

    // 6) output projection (WMMA) -> attn_output
    gemm_bf16<<<dim3(DMODEL / 64, MROWS / 32), 256, 0, stream>>>(Ob, Wob, attn_out, MROWS, DMODEL, NH * HD);
}
